// GraphAttentionLayer_69518340653442
// MI455X (gfx1250) — matrix operand order repaired and checked by reading, not compiled, not run on hardware
//
#include <hip/hip_runtime.h>
#include <hip/hip_bf16.h>
#include <math.h>

// ---- problem sizes (fixed by the reference) ----
#define N_NODES 8192
#define FIN     2048
#define FOUT    256
#define NCAM    8
#define PERM_CAP 8320            // 8192 + 8*16 padding headroom
#define ROW_TILES (PERM_CAP/16)  // 520
#define GAT_ALPHA 0.2f

typedef __bf16 v16bf __attribute__((ext_vector_type(16)));
typedef float  v8f   __attribute__((ext_vector_type(8)));

union FragAB { v16bf v; uint4 q[2]; };
union FragC  { v8f   v; float f[8]; };

__device__ __forceinline__ unsigned short f2bf(float f) {
    union { float f; unsigned u; } c; c.f = f;
    unsigned u = c.u;
    u += 0x7FFFu + ((u >> 16) & 1u);   // round-to-nearest-even
    return (unsigned short)(u >> 16);
}
__device__ __forceinline__ float bf2f(unsigned short h) {
    union { unsigned u; float f; } c; c.u = ((unsigned)h) << 16;
    return c.f;
}

// ---------------- 0a: x f32 -> bf16 ----------------
__global__ void k_convert_x(const float* __restrict__ x, unsigned short* __restrict__ xb) {
    unsigned total = (unsigned)N_NODES * FIN;
    for (unsigned i = blockIdx.x * blockDim.x + threadIdx.x; i < total;
         i += gridDim.x * blockDim.x)
        xb[i] = f2bf(x[i]);
}

// ---------------- 0b: W (c,k,o) f32 -> Wt (c,o,k) bf16 (B-friendly layout) ----
__global__ void k_convert_w(const float* __restrict__ W, unsigned short* __restrict__ Wt) {
    unsigned total = (unsigned)NCAM * FIN * FOUT;
    for (unsigned i = blockIdx.x * blockDim.x + threadIdx.x; i < total;
         i += gridDim.x * blockDim.x) {
        unsigned c = i / (FIN * FOUT);
        unsigned r = i % (FIN * FOUT);
        unsigned k = r / FOUT, o = r % FOUT;
        Wt[((size_t)c * FOUT + o) * FIN + k] = f2bf(W[i]);
    }
}

// ---------------- 1: bucket nodes by camera, pad groups to 16 ----------------
// Also zeroes the dummy A-row used by padded perm slots.
__global__ void k_bucket(const int* __restrict__ cam, int* __restrict__ perm,
                         unsigned short* __restrict__ zrow) {
    __shared__ int cnt[NCAM];
    __shared__ int cur[NCAM];
    int t = threadIdx.x;
    if (t < NCAM) cnt[t] = 0;
    __syncthreads();
    for (int n = t; n < N_NODES; n += blockDim.x) atomicAdd(&cnt[cam[n]], 1);
    __syncthreads();
    if (t == 0) {
        int o = 0;
        for (int c = 0; c < NCAM; ++c) { cur[c] = o; o += (cnt[c] + 15) & ~15; }
    }
    for (int p = t; p < PERM_CAP; p += blockDim.x) perm[p] = -1;
    for (int k = t; k < FIN; k += blockDim.x) zrow[k] = 0;
    __syncthreads();
    for (int n = t; n < N_NODES; n += blockDim.x) {
        int c = cam[n];
        int pos = atomicAdd(&cur[c], 1);
        perm[pos] = n;
    }
}

// ---------------- 2: h = x @ W[cam]  (bf16 WMMA), store transposed hT[o][n] ---
__global__ void __launch_bounds__(256)
k_h_gemm(const unsigned short* __restrict__ xb, const unsigned short* __restrict__ Wt,
         const int* __restrict__ perm, const int* __restrict__ cam,
         const unsigned short* __restrict__ zrow, unsigned short* __restrict__ hT) {
    const int wave = threadIdx.x >> 5;
    const int lane = threadIdx.x & 31;
    const int l15  = lane & 15;
    const int hi   = lane >> 4;
    const int rt   = blockIdx.x;

    const int nodeA  = perm[rt * 16 + l15];
    const int anchor = perm[rt * 16];
    const int c      = (anchor >= 0) ? cam[anchor] : 0;   // tile is single-camera by construction

    // padded lanes read a pre-zeroed dummy row: no predication in the hot loop
    const unsigned short* arow =
        (nodeA >= 0 ? xb + (size_t)nodeA * FIN : zrow) + hi * 8;
    const int o0 = wave * 32;
    const unsigned short* b0 =
        Wt + ((size_t)c * FOUT + o0 + l15) * FIN + hi * 8;
    const unsigned short* b1 = b0 + (size_t)16 * FIN;

    FragC acc0, acc1;
    #pragma unroll
    for (int e = 0; e < 8; ++e) { acc0.f[e] = 0.f; acc1.f[e] = 0.f; }

    for (int k = 0; k < FIN; k += 32) {
        FragAB fa;
        fa.q[0] = *(const uint4*)(arow + k);        // K = k+kb   .. +7
        fa.q[1] = *(const uint4*)(arow + k + 16);   // K = k+kb+16.. +23
        FragAB fb0, fb1;
        fb0.q[0] = *(const uint4*)(b0 + k);
        fb0.q[1] = *(const uint4*)(b0 + k + 16);
        fb1.q[0] = *(const uint4*)(b1 + k);
        fb1.q[1] = *(const uint4*)(b1 + k + 16);
        acc0.v = __builtin_amdgcn_wmma_f32_16x16x32_bf16(false, fa.v, false, fb0.v,
                                                         (short)0, acc0.v, false, false);
        acc1.v = __builtin_amdgcn_wmma_f32_16x16x32_bf16(false, fa.v, false, fb1.v,
                                                         (short)0, acc1.v, false, false);
    }

    // D layout: VGPR e holds M = e + 8*hi, N = l15. Scatter through perm into hT.
    #pragma unroll
    for (int e = 0; e < 8; ++e) {
        int m = e + hi * 8;
        int p = perm[rt * 16 + m];
        if (p >= 0) {
            hT[(size_t)(o0 + l15)      * N_NODES + p] = f2bf(acc0.f[e]);
            hT[(size_t)(o0 + 16 + l15) * N_NODES + p] = f2bf(acc1.f[e]);
        }
    }
}

// ---------------- 3: s1 = h@a1, s2 = h@a2 ----------------
__global__ void k_scores(const unsigned short* __restrict__ hT, const float* __restrict__ a,
                         float* __restrict__ s1, float* __restrict__ s2) {
    int n = blockIdx.x * blockDim.x + threadIdx.x;
    if (n >= N_NODES) return;
    float acc1 = 0.f, acc2 = 0.f;
    for (int o = 0; o < FOUT; ++o) {
        float h = bf2f(hT[(size_t)o * N_NODES + n]);   // coalesced across threads
        acc1 += h * a[o];
        acc2 += h * a[FOUT + o];
    }
    s1[n] = acc1; s2[n] = acc2;
}

// ---------------- 4: masked row max (leaky_relu is monotone: max over s2) ----
__global__ void __launch_bounds__(256)
k_rowmax(const int* __restrict__ adj, const float* __restrict__ s1,
         const float* __restrict__ s2, float* __restrict__ mrow) {
    int wave = threadIdx.x >> 5, lane = threadIdx.x & 31;
    int i = blockIdx.x * 8 + wave;
    const int* row = adj + (size_t)i * N_NODES;
    float sm = -INFINITY;
    for (int j = lane; j < N_NODES; j += 32)
        if (row[j] > 0) sm = fmaxf(sm, s2[j]);
    for (int off = 16; off; off >>= 1) sm = fmaxf(sm, __shfl_xor(sm, off, 32));
    if (lane == 0) {
        float t = s1[i] + sm;
        mrow[i] = (t > 0.f) ? t : GAT_ALPHA * t;
    }
}

// ---------------- 5: fused softmax + att@h + elu (WMMA, P-tile via LDS) ------
__global__ void __launch_bounds__(256)
k_att(const int* __restrict__ adj, const float* __restrict__ s1,
      const float* __restrict__ s2, const float* __restrict__ mrow,
      const unsigned short* __restrict__ hT, float* __restrict__ out) {
    __shared__ __align__(16) unsigned short P[16 * 32];  // 16x32 bf16 prob tile
    __shared__ float lsum[16];

    const int tid  = threadIdx.x;
    const int wave = tid >> 5, lane = tid & 31;
    const int l15  = lane & 15, hi = lane >> 4;
    const int i0   = blockIdx.x * 16;

    // each thread owns two fixed (row, jcol) P slots for the whole K sweep
    const int rA = wave, rB = 8 + wave, jc = lane;
    const float s1A = s1[i0 + rA], mA = mrow[i0 + rA];
    const float s1B = s1[i0 + rB], mB = mrow[i0 + rB];
    float lpA = 0.f, lpB = 0.f;

    const int o0 = wave * 32;
    const unsigned short* b0 = hT + (size_t)(o0 + l15) * N_NODES + hi * 8;
    const unsigned short* b1 = b0 + (size_t)16 * N_NODES;

    FragC acc0, acc1;
    #pragma unroll
    for (int e = 0; e < 8; ++e) { acc0.f[e] = 0.f; acc1.f[e] = 0.f; }

    const int* rowAp = adj + (size_t)(i0 + rA) * N_NODES;
    const int* rowBp = adj + (size_t)(i0 + rB) * N_NODES;

    for (int j0 = 0; j0 < N_NODES; j0 += 32) {
        // build probability tile: exp computed exactly once per (i,j)
        {
            float s2j = s2[j0 + jc];
            float tA = s1A + s2j;
            float lA = tA > 0.f ? tA : GAT_ALPHA * tA;
            float pA = (rowAp[j0 + jc] > 0) ? __expf(lA - mA) : 0.f;
            lpA += pA;
            P[rA * 32 + jc] = f2bf(pA);
            float tB = s1B + s2j;
            float lB = tB > 0.f ? tB : GAT_ALPHA * tB;
            float pB = (rowBp[j0 + jc] > 0) ? __expf(lB - mB) : 0.f;
            lpB += pB;
            P[rB * 32 + jc] = f2bf(pB);
        }
        if (j0 + 1024 < N_NODES) {
            __builtin_prefetch(&rowAp[j0 + 1024], 0, 1);   // global_prefetch_b8
            __builtin_prefetch(&rowBp[j0 + 1024], 0, 1);
        }
        __syncthreads();

        // A fragment straight from LDS (row = l15, K runs kb..kb+7, kb+16..kb+23)
        FragAB fa;
        const unsigned short* pl = P + l15 * 32 + hi * 8;
        fa.q[0] = *(const uint4*)(pl);        // elements K: kb .. kb+7
        fa.q[1] = *(const uint4*)(pl + 16);   // elements K: kb+16 .. kb+23

        FragAB fb0, fb1;
        fb0.q[0] = *(const uint4*)(b0 + j0);
        fb0.q[1] = *(const uint4*)(b0 + j0 + 16);
        fb1.q[0] = *(const uint4*)(b1 + j0);
        fb1.q[1] = *(const uint4*)(b1 + j0 + 16);

        acc0.v = __builtin_amdgcn_wmma_f32_16x16x32_bf16(false, fa.v, false, fb0.v,
                                                         (short)0, acc0.v, false, false);
        acc1.v = __builtin_amdgcn_wmma_f32_16x16x32_bf16(false, fa.v, false, fb1.v,
                                                         (short)0, acc1.v, false, false);
        __syncthreads();
    }

    // row sums (softmax denominators): wave w owns rows w and w+8
    float rsA = lpA, rsB = lpB;
    for (int off = 16; off; off >>= 1) {
        rsA += __shfl_xor(rsA, off, 32);
        rsB += __shfl_xor(rsB, off, 32);
    }
    if (lane == 0) { lsum[rA] = rsA; lsum[rB] = rsB; }
    __syncthreads();

    // normalize + elu + store
    #pragma unroll
    for (int e = 0; e < 8; ++e) {
        int m = e + hi * 8;
        float inv = 1.0f / lsum[m];
        float v0 = acc0.f[e] * inv;
        float v1 = acc1.f[e] * inv;
        v0 = v0 > 0.f ? v0 : (__expf(v0) - 1.f);
        v1 = v1 > 0.f ? v1 : (__expf(v1) - 1.f);
        out[(size_t)(i0 + m) * FOUT + o0 + l15]      = v0;
        out[(size_t)(i0 + m) * FOUT + o0 + 16 + l15] = v1;
    }
}

// ---------------- host-side orchestration ----------------
extern "C" void kernel_launch(void* const* d_in, const int* in_sizes, int n_in,
                              void* d_out, int out_size, void* d_ws, size_t ws_size,
                              hipStream_t stream) {
    const float* x   = (const float*)d_in[0];
    const float* W   = (const float*)d_in[1];
    const float* a   = (const float*)d_in[2];
    const int*   cam = (const int*)d_in[3];
    const int*   adj = (const int*)d_in[4];
    float* out = (float*)d_out;

    char* ws = (char*)d_ws;
    unsigned short* xb = (unsigned short*)(ws);                       // 33,554,432 B
    unsigned short* Wt = (unsigned short*)(ws + 33554432);            //  8,388,608 B
    unsigned short* hT = (unsigned short*)(ws + 33554432 + 8388608);  //  4,194,304 B
    char* tail = ws + 33554432 + 8388608 + 4194304;
    int*   perm = (int*)(tail);                                       // 33,280 B (pad to 36,864)
    float* s1   = (float*)(tail + 36864);
    float* s2   = s1 + N_NODES;
    float* mrow = s2 + N_NODES;
    unsigned short* zrow = (unsigned short*)(mrow + N_NODES);         // FIN bf16 zeros (4 KB)

    k_convert_x<<<2048, 256, 0, stream>>>(x, xb);
    k_convert_w<<<2048, 256, 0, stream>>>(W, Wt);
    k_bucket<<<1, 1024, 0, stream>>>(cam, perm, zrow);
    k_h_gemm<<<ROW_TILES, 256, 0, stream>>>(xb, Wt, perm, cam, zrow, hT);
    k_scores<<<(N_NODES + 255) / 256, 256, 0, stream>>>(hT, a, s1, s2);
    k_rowmax<<<N_NODES / 8, 256, 0, stream>>>(adj, s1, s2, mrow);     // 1024 blocks, wave per row
    k_att<<<N_NODES / 16, 256, 0, stream>>>(adj, s1, s2, mrow, hT, out);
}
